// Attention_GRU_70291434766493
// MI455X (gfx1250) — compile-verified, matrix operand order checked by reading, not run on hardware
//
#include <hip/hip_runtime.h>
#include <math.h>

// ---------------------------------------------------------------------------
// Attention-GRU for MI455X (gfx1250, wave32):
//   * bf16 WMMA 16x16x32 (f32 accum) for both GEMMs
//   * TDM tensor_load_to_lds stages the shared A panel in LDS
//   * 4-wide N register blocking in the big input-projection GEMM
// ---------------------------------------------------------------------------

typedef __attribute__((ext_vector_type(16))) __bf16 v16bf;
typedef __attribute__((ext_vector_type(8)))  float  v8f;
typedef __attribute__((ext_vector_type(4)))  unsigned v4u;
typedef __attribute__((ext_vector_type(8)))  int   v8i;
typedef __attribute__((ext_vector_type(4)))  int   v4i;

constexpr int kB  = 32;    // batch
constexpr int kT  = 512;   // time steps
constexpr int kD  = 512;   // input dim
constexpr int kU  = 512;   // hidden units
constexpr int kTa = 256;   // attention rows
constexpr int kN3 = 3 * kU;  // 1536

// ---------------------------------------------------------------------------
// Helpers
// ---------------------------------------------------------------------------

__device__ inline v16bf ld_frag(const __bf16* p) {
  // Two aligned 16-byte loads: K-chunks [0..7] and [16..23] relative to hi*8.
  union { uint4 u[2]; v16bf v; } f;
  f.u[0] = *reinterpret_cast<const uint4*>(p);
  f.u[1] = *reinterpret_cast<const uint4*>(p + 16);
  return f.v;
}

__device__ inline v8f wmma_bf16(v16bf a, v16bf b, v8f c) {
  return __builtin_amdgcn_wmma_f32_16x16x32_bf16(
      /*neg_a=*/false, a, /*neg_b=*/false, b,
      /*c_mod=*/(short)0, c, /*reuse_a=*/false, /*reuse_b=*/false);
}

__device__ inline void store_tile(float* __restrict__ C, int ldc,
                                  int row0, int col0, v8f acc, int lane) {
  const int r  = lane & 15;
  const int hi = (lane >> 4) & 1;
#pragma unroll
  for (int v = 0; v < 8; ++v)
    C[(size_t)(row0 + v + hi * 8) * ldc + col0 + r] = acc[v];
}

__device__ inline float hsig(float v) {
  return fminf(fmaxf(0.2f * v + 0.5f, 0.0f), 1.0f);
}

// ---------------------------------------------------------------------------
// Tensor Data Mover: async 2-D tile load (global -> LDS), bf16 elements.
// D# layout per CDNA5 ISA ch.8: group0 = {count, lds_addr, global_addr, type},
// group1 = {data_size, tensor dims, tile dims, row stride}; groups 2/3 zero
// (2-D tensor). Issued by one wave; completion via s_wait_tensorcnt.
// This toolchain's builtin takes 6 args (groups 0..3 as 4+8+4+4 dwords, an
// extra 8-dword block, then cache-policy).
// ---------------------------------------------------------------------------

__device__ inline void tdm_load_2d_bf16(unsigned lds_off, const __bf16* gsrc,
                                        unsigned rows, unsigned row_elems) {
  unsigned long long ga = (unsigned long long)(uintptr_t)gsrc;
  v4u g0;
  g0[0] = 1u;                                   // count=1, user descriptor
  g0[1] = lds_off;                              // LDS byte address
  g0[2] = (unsigned)ga;                         // global_addr[31:0]
  g0[3] = (unsigned)((ga >> 32) & 0x01FFFFFFu)  // global_addr[56:32]
          | (2u << 30);                         // type=2 ("image")
  v8i g1;
  g1[0] = (int)0x00010000u;                     // data_size=1 (2 bytes)
  g1[1] = (int)((row_elems & 0xFFFFu) << 16);               // tensor_dim0 lo
  g1[2] = (int)(((row_elems >> 16) & 0xFFFFu)
          | ((rows & 0xFFFFu) << 16));                      // dim0 hi | dim1 lo
  g1[3] = (int)(((rows >> 16) & 0xFFFFu)
          | ((row_elems & 0xFFFFu) << 16));                 // dim1 hi | tile_dim0
  g1[4] = (int)(rows & 0xFFFFu);                            // tile_dim1
  g1[5] = (int)row_elems;                                   // dim0_stride lo32
  g1[6] = 0;                                                // stride hi bits
  g1[7] = 0;
  v4i z4;  z4[0] = 0; z4[1] = 0; z4[2] = 0; z4[3] = 0;
  v8i z8;
#pragma unroll
  for (int i = 0; i < 8; ++i) z8[i] = 0;
  __builtin_amdgcn_tensor_load_to_lds(g0, g1, z4, z4, z8, 0);
}

// ---------------------------------------------------------------------------
// Conversion / init kernels
// ---------------------------------------------------------------------------

__global__ __launch_bounds__(256) void cvt_bf16(const float* __restrict__ in,
                                                __bf16* __restrict__ out, int n) {
  for (int i = blockIdx.x * blockDim.x + threadIdx.x; i < n;
       i += gridDim.x * blockDim.x)
    out[i] = (__bf16)in[i];
}

// in: (K, N) fp32 row-major  ->  out: (N, K) bf16 row-major (transpose+convert)
__global__ __launch_bounds__(256) void cvt_T_bf16(const float* __restrict__ in,
                                                  __bf16* __restrict__ out,
                                                  int K, int N) {
  int idx = blockIdx.x * blockDim.x + threadIdx.x;
  if (idx >= K * N) return;
  int n = idx / K, k = idx - n * K;
  out[(size_t)n * K + k] = (__bf16)in[(size_t)k * N + n];
}

__global__ __launch_bounds__(256) void init_h(float* __restrict__ hf,
                                              __bf16* __restrict__ hb, int n) {
  int i = blockIdx.x * blockDim.x + threadIdx.x;
  if (i < n) { hf[i] = 0.0f; hb[i] = (__bf16)0.0f; }
}

// ---------------------------------------------------------------------------
// xw = x @ kernel + bias : (16384 x 512) @ (512 x 1536) -> fp32 (16384 x 1536)
// Block = 8 waves; all share one 16-row A panel, TDM-staged in LDS (16 KB).
// Each wave computes a 16x64 strip (4 N-tiles) -> one A fragment feeds 4
// WMMAs per K-step. Grid = (M/16) * (N/512) = 1024 * 3 blocks.
// ---------------------------------------------------------------------------

__global__ __launch_bounds__(256) void gemm_xw_wmma(
    const __bf16* __restrict__ xb,   // (B*T, D) bf16
    const __bf16* __restrict__ kbT,  // (3U, D)  bf16 (kernel transposed)
    const float*  __restrict__ bias, // (3U)
    float* __restrict__ xw) {        // (B*T, 3U) fp32
  __shared__ alignas(16) __bf16 shA[16 * kD];  // 16 KB A panel

  const int wave = threadIdx.x >> 5;
  const int lane = threadIdx.x & 31;
  const int tile_m = blockIdx.x / 3;        // 0..1023
  const int blkN   = blockIdx.x % 3;        // 0..2 (512 cols each)
  const int r  = lane & 15;
  const int hi = (lane >> 4) & 1;

  if (wave == 0) {
    tdm_load_2d_bf16((unsigned)(uintptr_t)shA, xb + (size_t)tile_m * 16 * kD,
                     16, kD);
    __builtin_amdgcn_s_wait_tensorcnt(0);
  }
  __syncthreads();

  const int tn0 = blkN * 32 + wave * 4;     // first of 4 N-tiles for this wave
  v8f acc[4];
#pragma unroll
  for (int j = 0; j < 4; ++j) {
    const float bv = bias[(tn0 + j) * 16 + r];
#pragma unroll
    for (int v = 0; v < 8; ++v) acc[j][v] = bv;
  }

  const __bf16* ap = shA + (size_t)r * kD + hi * 8;  // LDS A fragment base
  const __bf16* bp[4];
#pragma unroll
  for (int j = 0; j < 4; ++j)
    bp[j] = kbT + (size_t)((tn0 + j) * 16 + r) * kD + hi * 8;

#pragma unroll 2
  for (int k0 = 0; k0 < kD; k0 += 32) {
    const v16bf av = ld_frag(ap + k0);
#pragma unroll
    for (int j = 0; j < 4; ++j)
      acc[j] = wmma_bf16(av, ld_frag(bp[j] + k0), acc[j]);
  }

#pragma unroll
  for (int j = 0; j < 4; ++j)
    store_tile(xw, kN3, tile_m * 16, (tn0 + j) * 16, acc[j], lane);
}

// ---------------------------------------------------------------------------
// g = h @ recurrent_kernel : (32 x 512) @ (512 x 1536) -> fp32 (32 x 1536)
// On the 512-step critical path. The whole h (32x512 bf16 = 32 KB) is TDM-
// staged in LDS per block; 24 blocks x 8 waves, one 16x16 tile per wave for
// maximum parallelism at tiny M.
// ---------------------------------------------------------------------------

__global__ __launch_bounds__(256) void gemm_gates_wmma(
    const __bf16* __restrict__ hb,   // (32, U) bf16
    const __bf16* __restrict__ UbT,  // (3U, U) bf16 (recurrent transposed)
    float* __restrict__ g) {         // (32, 3U) fp32
  __shared__ alignas(16) __bf16 shH[kB * kU];  // 32 KB: all of h

  const int wave = threadIdx.x >> 5;
  const int lane = threadIdx.x & 31;
  const int id = blockIdx.x * 8 + wave;  // 0..191
  const int tile_m = id / 96;            // 0..1
  const int tile_n = id % 96;            // 0..95
  const int r  = lane & 15;
  const int hi = (lane >> 4) & 1;

  if (wave == 0) {
    tdm_load_2d_bf16((unsigned)(uintptr_t)shH, hb, kB, kU);
    __builtin_amdgcn_s_wait_tensorcnt(0);
  }
  __syncthreads();

  v8f acc;
#pragma unroll
  for (int v = 0; v < 8; ++v) acc[v] = 0.0f;

  const __bf16* ap = shH + (size_t)(tile_m * 16 + r) * kU + hi * 8;
  const __bf16* bp = UbT + (size_t)(tile_n * 16 + r) * kU + hi * 8;
#pragma unroll 4
  for (int k0 = 0; k0 < kU; k0 += 32)
    acc = wmma_bf16(ld_frag(ap + k0), ld_frag(bp + k0), acc);

  store_tile(g, kN3, tile_m * 16, tile_n * 16, acc, lane);
}

// ---------------------------------------------------------------------------
// Fused GRU gate update + attention readout for one time step.
// One block per batch element (32 blocks x 256 threads). att (16 MB fp32) is
// fully L2-resident on MI455X (192 MB L2); h and a live in LDS.
// ---------------------------------------------------------------------------

__global__ __launch_bounds__(256) void gru_step_attn(
    const float* __restrict__ xw,   // (B*T, 3U)
    const float* __restrict__ g,    // (B, 3U)
    const float* __restrict__ att,  // (B, Ta, U)
    float* __restrict__ hf,         // (B, U) fp32 state
    __bf16* __restrict__ hb,        // (B, U) bf16 state (for next gates GEMM)
    float* __restrict__ out,        // (B, T, U)
    int t) {
  __shared__ float h_sh[kU];
  __shared__ float a_sh[kTa];

  const int b = blockIdx.x;
  const int tid = threadIdx.x;
  const float* xwrow = xw + ((size_t)b * kT + t) * kN3;
  const float* grow  = g + (size_t)b * kN3;

  // --- gate update: each thread handles u and u+256 ---
#pragma unroll
  for (int j = 0; j < 2; ++j) {
    const int u = tid + j * 256;
    const float z  = hsig(xwrow[u] + grow[u]);
    const float r  = hsig(xwrow[kU + u] + grow[kU + u]);
    const float hh = tanhf(xwrow[2 * kU + u] + r * grow[2 * kU + u]);
    const float hp = hf[(size_t)b * kU + u];
    const float hn = z * hp + (1.0f - z) * hh;
    hf[(size_t)b * kU + u] = hn;
    hb[(size_t)b * kU + u] = (__bf16)hn;
    h_sh[u] = hn;
  }
  __syncthreads();

  // --- a[s] = <h, att[b,s,:]> ; one s per thread (contiguous reads) ---
  {
    const int s = tid;
    const float* arow = att + ((size_t)b * kTa + s) * kU;
    float acc = 0.0f;
#pragma unroll 4
    for (int u = 0; u < kU; ++u) acc += h_sh[u] * arow[u];
    a_sh[s] = acc;
  }
  __syncthreads();

  // --- o[u] = sum_s a[s] * att[b,s,u] ; strided over s, coalesced over u ---
#pragma unroll
  for (int j = 0; j < 2; ++j) {
    const int u = tid + j * 256;
    const float* ab = att + (size_t)b * kTa * kU + u;
    float acc = 0.0f;
    for (int s = 0; s < kTa; ++s) {
      if ((s & 7) == 0 && s + 8 < kTa)
        __builtin_prefetch(ab + (size_t)(s + 8) * kU, 0, 1);
      acc += a_sh[s] * ab[(size_t)s * kU];
    }
    out[((size_t)b * kT + t) * kU + u] = acc;
  }
}

// ---------------------------------------------------------------------------
// Host launcher: deterministic, all on `stream`, graph-capture safe.
// d_in order: x, att, kernel, recurrent_kernel, bias
// ---------------------------------------------------------------------------

extern "C" void kernel_launch(void* const* d_in, const int* in_sizes, int n_in,
                              void* d_out, int out_size, void* d_ws, size_t ws_size,
                              hipStream_t stream) {
  const float* x    = (const float*)d_in[0];
  const float* att  = (const float*)d_in[1];
  const float* kern = (const float*)d_in[2];
  const float* rker = (const float*)d_in[3];
  const float* bias = (const float*)d_in[4];
  float* out = (float*)d_out;

  // Workspace carve-up (all chunks 256B-aligned sizes).
  char* ws = (char*)d_ws;
  size_t off = 0;
  __bf16* xb  = (__bf16*)(ws + off); off += (size_t)kB * kT * kD * sizeof(__bf16);
  __bf16* kbT = (__bf16*)(ws + off); off += (size_t)kN3 * kD * sizeof(__bf16);
  __bf16* UbT = (__bf16*)(ws + off); off += (size_t)kN3 * kU * sizeof(__bf16);
  float*  xw  = (float*)(ws + off);  off += (size_t)kB * kT * kN3 * sizeof(float);
  float*  g   = (float*)(ws + off);  off += (size_t)kB * kN3 * sizeof(float);
  float*  hf  = (float*)(ws + off);  off += (size_t)kB * kU * sizeof(float);
  __bf16* hb  = (__bf16*)(ws + off); off += (size_t)kB * kU * sizeof(__bf16);

  // 1) bf16 conversions (weights transposed to N-major so WMMA B fragments
  //    are contiguous over K).
  cvt_bf16<<<4096, 256, 0, stream>>>(x, xb, kB * kT * kD);
  cvt_T_bf16<<<(kD * kN3 + 255) / 256, 256, 0, stream>>>(kern, kbT, kD, kN3);
  cvt_T_bf16<<<(kU * kN3 + 255) / 256, 256, 0, stream>>>(rker, UbT, kU, kN3);
  init_h<<<(kB * kU + 255) / 256, 256, 0, stream>>>(hf, hb, kB * kU);

  // 2) One-shot input projection: xw = x @ kernel + bias
  gemm_xw_wmma<<<(kB * kT / 16) * 3, 256, 0, stream>>>(xb, kbT, bias, xw);

  // 3) Sequential scan: per step a small WMMA GEMM + fused gates/attention.
  for (int t = 0; t < kT; ++t) {
    gemm_gates_wmma<<<24, 256, 0, stream>>>(hb, UbT, g);
    gru_step_attn<<<kB, 256, 0, stream>>>(xw, g, att, hf, hb, out, t);
  }
}
